// CANNAttention_37898791419999
// MI455X (gfx1250) — matrix operand order repaired and checked by reading, not compiled, not run on hardware
//
#include <hip/hip_runtime.h>
#include <hip/hip_bf16.h>

#define H_  1024
#define NH_ 16
#define HD_ 64
#define B_  2
#define S_  2048
#define M_  (B_*S_)   // 4096

typedef __attribute__((ext_vector_type(16))) __bf16    v16bf;
typedef __attribute__((ext_vector_type(8)))  __bf16    v8bf;
typedef __attribute__((ext_vector_type(8)))  float     v8f;
typedef __attribute__((ext_vector_type(4)))  unsigned  uint32x4;
typedef __attribute__((ext_vector_type(8)))  int       int32x8;
typedef __attribute__((ext_vector_type(4)))  int       int32x4;

static __device__ inline v8f wmma_bf16(v16bf a, v16bf b, v8f c) {
  // D = A(16x32 bf16) x B(32x16 bf16) + C(16x16 f32)
  return __builtin_amdgcn_wmma_f32_16x16x32_bf16(false, a, false, b, (short)0, c,
                                                 false, false);
}

// A-fragment: 16 contiguous elements at p (two 16B chunks at p and p+8).
static __device__ inline v16bf load_a16(const __bf16* p) {
  v8bf lo = *(const v8bf*)p;
  v8bf hi = *(const v8bf*)(p + 8);
  v16bf v;
  #pragma unroll
  for (int j = 0; j < 8; ++j) { v[j] = lo[j]; v[8 + j] = hi[j]; }
  return v;
}

// ---------------------------------------------------------------------------
// TDM: DMA a 2-D bf16 tile (tile_d0 x tile_d1 elems, row stride stride0 elems)
// from global memory into LDS.  Descriptor bit layout per CDNA5 ISA §8.3-8.6.
// data_size code 1 = 2 bytes/elem.  Tracked by TENSORcnt.
// 6-arg builtin form (clang-23): (g0, g1, g2, g3, g4, cpol).
// ---------------------------------------------------------------------------
static __device__ inline void tdm_load_2d(unsigned lds_byte_addr,
                                          const __bf16* gaddr,
                                          unsigned tensor_d0, unsigned tensor_d1,
                                          unsigned tile_d0,   unsigned tile_d1,
                                          unsigned stride0) {
  unsigned long long ga = (unsigned long long)gaddr;
  uint32x4 g0;
  g0[0] = 1u;                                            // count=1, user D#
  g0[1] = lds_byte_addr;                                 // lds_addr [63:32]
  g0[2] = (unsigned)(ga & 0xffffffffu);                  // global_addr lo
  g0[3] = (unsigned)((ga >> 32) & 0x01ffffffu)           // global_addr hi (57b)
        | (2u << 30);                                    // type=2 ("image")
  int32x8 g1;
  g1[0] = (int)(1u << 16);                               // wg_mask=0, data_size=1(2B)
  g1[1] = (int)((tensor_d0 & 0xffffu) << 16);            // tensor_dim0[15:0] @63:48
  g1[2] = (int)(((tensor_d0 >> 16) & 0xffffu)            // tensor_dim0[31:16]
        | ((tensor_d1 & 0xffffu) << 16));                // tensor_dim1[15:0]
  g1[3] = (int)(((tensor_d1 >> 16) & 0xffffu)            // tensor_dim1[31:16]
        | ((tile_d0 & 0xffffu) << 16));                  // tile_dim0 @127:112
  g1[4] = (int)(tile_d1 & 0xffffu);                      // tile_dim1; tile_dim2=0
  g1[5] = (int)stride0;                                  // tensor_dim0_stride lo32
  g1[6] = 0;                                             // stride0 hi16, dim1_stride lo
  g1[7] = 0;
  int32x4 zz = {0, 0, 0, 0};                             // groups 2/3 unused (2-D)
  int32x8 z8 = {0, 0, 0, 0, 0, 0, 0, 0};                 // extra group (unused)
  __builtin_amdgcn_tensor_load_to_lds(g0, g1, zz, zz, z8, 0);
}

// ---------------------------------------------------------------------------
// Kernel 0: f32 -> bf16 conversion pass (bandwidth-bound, 8 elems/thread).
// ---------------------------------------------------------------------------
__global__ __launch_bounds__(256) void cvt_bf16_kernel(
    const float* __restrict__ src, __bf16* __restrict__ dst, int n)
{
  int i = (blockIdx.x * 256 + threadIdx.x) * 8;
  if (i >= n) return;
  float4 f0 = *(const float4*)(src + i);
  float4 f1 = *(const float4*)(src + i + 4);
  v8bf o;
  o[0] = (__bf16)f0.x; o[1] = (__bf16)f0.y; o[2] = (__bf16)f0.z; o[3] = (__bf16)f0.w;
  o[4] = (__bf16)f1.x; o[5] = (__bf16)f1.y; o[6] = (__bf16)f1.z; o[7] = (__bf16)f1.w;
  *(v8bf*)(dst + i) = o;
}

// ---------------------------------------------------------------------------
// Kernel 1: QKV projection (bf16 x bf16 -> f32 acc).  y = x @ W^T + b.
// z=0 -> Q (scaled 1/8) into [bh, s, d]; z=1 -> K [bh, s, d];
// z=2 -> V transposed [bh, d, s].
// Block = 256 threads (8 waves), WG tile 64x128, wave tile 16x64.
// ---------------------------------------------------------------------------
__global__ __launch_bounds__(256) void qkv_kernel(
    const __bf16* __restrict__ x, const __bf16* __restrict__ w_all,
    const float* __restrict__ bq, const float* __restrict__ bk,
    const float* __restrict__ bv,
    __bf16* __restrict__ q_ws, __bf16* __restrict__ k_ws,
    __bf16* __restrict__ vt_ws)
{
  const int z = blockIdx.z;
  const __bf16* W    = w_all + (size_t)z * H_ * H_;
  const float*  bias = (z == 0) ? bq : (z == 1) ? bk : bv;

  const int wave = threadIdx.x >> 5;
  const int lane = threadIdx.x & 31;
  const int half = lane >> 4;
  const int l16  = lane & 15;

  const int mBase = blockIdx.x * 64  + (wave & 3) * 16;
  const int nBase = blockIdx.y * 128 + (wave >> 2) * 64;

  const __bf16* ap = x + (size_t)(mBase + l16) * H_ + half * 16;  // A chunks: +0,+8
  const __bf16* bp = W + (size_t)(nBase + l16) * H_ + half * 16;  // B tiles: +t*16*H_

  v8f acc[4] = {};
  v16bf a  = load_a16(ap);
  v16bf b0 = *(const v16bf*)(bp);
  v16bf b1 = *(const v16bf*)(bp + 16 * H_);
  v16bf b2 = *(const v16bf*)(bp + 32 * H_);
  v16bf b3 = *(const v16bf*)(bp + 48 * H_);

  for (int kk = 0; kk < H_; kk += 32) {
    const int kn = (kk + 32 < H_) ? kk + 32 : kk;   // clamped prefetch
    v16bf an  = load_a16(ap + kn);
    v16bf b0n = *(const v16bf*)(bp + kn);
    v16bf b1n = *(const v16bf*)(bp + 16 * H_ + kn);
    v16bf b2n = *(const v16bf*)(bp + 32 * H_ + kn);
    v16bf b3n = *(const v16bf*)(bp + 48 * H_ + kn);
    acc[0] = wmma_bf16(a, b0, acc[0]);
    acc[1] = wmma_bf16(a, b1, acc[1]);
    acc[2] = wmma_bf16(a, b2, acc[2]);
    acc[3] = wmma_bf16(a, b3, acc[3]);
    a = an; b0 = b0n; b1 = b1n; b2 = b2n; b3 = b3n;
  }

  const float scale = (z == 0) ? 0.125f : 1.0f;   // fold 1/sqrt(HD) into Q
  #pragma unroll
  for (int t = 0; t < 4; ++t) {
    const int col = nBase + t * 16 + l16;
    const float bcol = bias[col];
    const int h = col >> 6, d = col & (HD_ - 1);
    #pragma unroll
    for (int r = 0; r < 8; ++r) {
      const int row = mBase + r + 8 * half;        // C layout: M = r + 8*half
      const float v = (acc[t][r] + bcol) * scale;
      const int bb = row >> 11, s = row & (S_ - 1);
      const size_t bh = (size_t)(bb * NH_ + h);
      if (z == 0)      q_ws [(bh * S_  + s) * HD_ + d] = (__bf16)v;
      else if (z == 1) k_ws [(bh * S_  + s) * HD_ + d] = (__bf16)v;
      else             vt_ws[(bh * HD_ + d) * S_  + s] = (__bf16)v;
    }
  }
}

// ---------------------------------------------------------------------------
// Kernel 2: streaming (flash) attention per (b,h) with TDM-fed LDS tiles.
// Block = 128 threads (4 waves); wave w owns 16 query rows.
// KV is consumed in 128-column chunks: wave 0 issues two tensor_load_to_lds
// DMAs per chunk (K tile 128x64, Vt tile 64x128) double-buffered, so chunk
// c+1 streams in while chunk c is computed (4 x [4 score WMMAs + softmax +
// 4 ctx WMMAs] per chunk, operands via ds_load_b128 from LDS).
// ---------------------------------------------------------------------------
__global__ __launch_bounds__(128) void attn_kernel(
    const __bf16* __restrict__ q_ws, const __bf16* __restrict__ k_ws,
    const __bf16* __restrict__ vt_ws, const float* __restrict__ mask,
    __bf16* __restrict__ ctx_ws)
{
  __shared__ __bf16 kbuf[2][128 * HD_];   // [kv 128][d 64]   16 KB x2
  __shared__ __bf16 vbuf[2][HD_ * 128];   // [d 64][kv 128]   16 KB x2
  __shared__ __bf16 p_lds[4][16 * 32];    // per-wave P re-layout staging

  const int wave = threadIdx.x >> 5;
  const int lane = threadIdx.x & 31;
  const int half = lane >> 4;
  const int l16  = lane & 15;

  const int bh = blockIdx.y;             // b*NH + h
  const int bb = bh >> 4;
  const int hh = bh & (NH_ - 1);
  const int qBase = blockIdx.x * 64 + wave * 16;

  const __bf16* Q  = q_ws  + (size_t)bh * S_ * HD_;
  const __bf16* Kg = k_ws  + (size_t)bh * S_ * HD_;
  const __bf16* Vg = vt_ws + (size_t)bh * HD_ * S_;
  const float*  mp = mask + (size_t)bb * S_ + l16;

  const unsigned kb_lds[2] = { (unsigned)(unsigned long long)(void*)&kbuf[0][0],
                               (unsigned)(unsigned long long)(void*)&kbuf[1][0] };
  const unsigned vb_lds[2] = { (unsigned)(unsigned long long)(void*)&vbuf[0][0],
                               (unsigned)(unsigned long long)(void*)&vbuf[1][0] };

  // Loop-invariant Q fragments covering d = 0..31 and 32..63.
  v16bf qa0, qa1;
  {
    const __bf16* qp = Q + (size_t)(qBase + l16) * HD_ + half * 16;
    qa0 = load_a16(qp);
    qa1 = load_a16(qp + 32);
  }

  float m[8], l[8];
  v8f acc[4] = {};
  #pragma unroll
  for (int r = 0; r < 8; ++r) { m[r] = -1e30f; l[r] = 0.f; }

  __bf16* myp = &p_lds[wave][0];
  const int nch = S_ / 128;

  // Kick off DMA of chunk 0 into buffer 0.
  if (wave == 0) {
    // K tile: rows kv 0..127, 64 cols, row-major stride HD.
    tdm_load_2d(kb_lds[0], Kg, HD_, S_, HD_, 128, HD_);
    // Vt tile: rows d 0..63, cols kv 0..127 of a [HD_, S_] tensor.
    tdm_load_2d(vb_lds[0], Vg, S_, HD_, 128, HD_, S_);
  }

  int buf = 0;
  for (int ch = 0; ch < nch; ++ch) {
    if (wave == 0) {
      if (ch + 1 < nch) {
        const int knext = (ch + 1) * 128;
        tdm_load_2d(kb_lds[buf ^ 1], Kg + (size_t)knext * HD_, HD_, S_, HD_, 128, HD_);
        tdm_load_2d(vb_lds[buf ^ 1], Vg + knext, S_, HD_, 128, HD_, S_);
        __builtin_amdgcn_s_wait_tensorcnt((short)2);  // chunk ch's 2 DMAs done
      } else {
        __builtin_amdgcn_s_wait_tensorcnt((short)0);
      }
    }
    __syncthreads();                                  // LDS tile visible to all

    const __bf16* kb = &kbuf[buf][0];
    const __bf16* vb = &vbuf[buf][0];
    const int kchunk = ch * 128;

    #pragma unroll
    for (int kt4 = 0; kt4 < 128; kt4 += 32) {
      // --- scores: two 16x16 tiles (kv cols +0..15, +16..31) ---
      v8f s[2];
      #pragma unroll
      for (int t = 0; t < 2; ++t) {
        const __bf16* kp = kb + (size_t)(kt4 + t * 16 + l16) * HD_ + half * 16;
        v16bf c0 = *(const v16bf*)(kp);
        v16bf c1 = *(const v16bf*)(kp + 32);
        v8f zc = {};
        zc = wmma_bf16(qa0, c0, zc);
        zc = wmma_bf16(qa1, c1, zc);
        const float mv = mp[kchunk + kt4 + t * 16];   // col-indexed mask
        #pragma unroll
        for (int r = 0; r < 8; ++r) zc[r] += mv;
        s[t] = zc;
      }

      // --- online softmax (row = r + 8*half; elems across 16-lane half) ---
      float p0[8], p1[8];
      #pragma unroll
      for (int r = 0; r < 8; ++r) {
        float tm = fmaxf(s[0][r], s[1][r]);
        #pragma unroll
        for (int msk = 8; msk >= 1; msk >>= 1)
          tm = fmaxf(tm, __shfl_xor(tm, msk, 32));
        const float mn = fmaxf(m[r], tm);
        const float alpha = __expf(m[r] - mn);
        const float e0 = __expf(s[0][r] - mn);
        const float e1 = __expf(s[1][r] - mn);
        float ts = e0 + e1;
        #pragma unroll
        for (int msk = 8; msk >= 1; msk >>= 1) ts += __shfl_xor(ts, msk, 32);
        l[r] = l[r] * alpha + ts;
        m[r] = mn;
        p0[r] = e0; p1[r] = e1;
        #pragma unroll
        for (int c = 0; c < 4; ++c) acc[c][r] *= alpha;
      }

      // --- C-layout -> A-layout transpose of P through per-wave LDS ---
      #pragma unroll
      for (int r = 0; r < 8; ++r) {
        myp[(r + 8 * half) * 32 + l16]      = (__bf16)p0[r];
        myp[(r + 8 * half) * 32 + 16 + l16] = (__bf16)p1[r];
      }
      asm volatile("s_wait_dscnt 0x0" ::: "memory");  // wave-local LDS RAW
      v16bf pa = load_a16(myp + l16 * 32 + half * 16);

      // --- ctx += P x V  (B-operand contiguous rows of Vt tile in LDS) ---
      #pragma unroll
      for (int c = 0; c < 4; ++c) {
        v16bf vv = *(const v16bf*)(vb + (size_t)(c * 16 + l16) * 128 + kt4 + half * 16);
        acc[c] = wmma_bf16(pa, vv, acc[c]);
      }
    }

    __syncthreads();    // all waves done with buf before its DMA is reissued
    buf ^= 1;
  }

  // normalize and emit ctx as bf16 in [M, H] row-major ([B,S,NH,HD] merged)
  #pragma unroll
  for (int r = 0; r < 8; ++r) {
    const float inv = 1.f / l[r];
    const int row = bb * S_ + qBase + r + 8 * half;
    #pragma unroll
    for (int c = 0; c < 4; ++c) {
      const int col = hh * HD_ + c * 16 + l16;
      ctx_ws[(size_t)row * H_ + col] = (__bf16)(acc[c][r] * inv);
    }
  }
}

// ---------------------------------------------------------------------------
// Kernel 3: output projection.  out = ctx @ wo^T + bo  (fp32 out).
// ---------------------------------------------------------------------------
__global__ __launch_bounds__(256) void outproj_kernel(
    const __bf16* __restrict__ ctx, const __bf16* __restrict__ wo_bf,
    const float* __restrict__ bo, float* __restrict__ out)
{
  const int wave = threadIdx.x >> 5;
  const int lane = threadIdx.x & 31;
  const int half = lane >> 4;
  const int l16  = lane & 15;
  const int mBase = blockIdx.x * 64  + (wave & 3) * 16;
  const int nBase = blockIdx.y * 128 + (wave >> 2) * 64;

  const __bf16* ap = ctx   + (size_t)(mBase + l16) * H_ + half * 16;
  const __bf16* bp = wo_bf + (size_t)(nBase + l16) * H_ + half * 16;

  v8f acc[4] = {};
  v16bf a  = load_a16(ap);
  v16bf b0 = *(const v16bf*)(bp);
  v16bf b1 = *(const v16bf*)(bp + 16 * H_);
  v16bf b2 = *(const v16bf*)(bp + 32 * H_);
  v16bf b3 = *(const v16bf*)(bp + 48 * H_);

  for (int kk = 0; kk < H_; kk += 32) {
    const int kn = (kk + 32 < H_) ? kk + 32 : kk;
    v16bf an  = load_a16(ap + kn);
    v16bf b0n = *(const v16bf*)(bp + kn);
    v16bf b1n = *(const v16bf*)(bp + 16 * H_ + kn);
    v16bf b2n = *(const v16bf*)(bp + 32 * H_ + kn);
    v16bf b3n = *(const v16bf*)(bp + 48 * H_ + kn);
    acc[0] = wmma_bf16(a, b0, acc[0]);
    acc[1] = wmma_bf16(a, b1, acc[1]);
    acc[2] = wmma_bf16(a, b2, acc[2]);
    acc[3] = wmma_bf16(a, b3, acc[3]);
    a = an; b0 = b0n; b1 = b1n; b2 = b2n; b3 = b3n;
  }

  #pragma unroll
  for (int t = 0; t < 4; ++t) {
    const int col = nBase + t * 16 + l16;
    const float bcol = bo[col];
    #pragma unroll
    for (int r = 0; r < 8; ++r) {
      const int row = mBase + r + 8 * half;
      out[(size_t)row * H_ + col] = acc[t][r] + bcol;
    }
  }
}

// ---------------------------------------------------------------------------
extern "C" void kernel_launch(void* const* d_in, const int* in_sizes, int n_in,
                              void* d_out, int out_size, void* d_ws, size_t ws_size,
                              hipStream_t stream) {
  const float* hs   = (const float*)d_in[0];
  const float* mask = (const float*)d_in[1];
  const float* wq = (const float*)d_in[2]; const float* bq = (const float*)d_in[3];
  const float* wk = (const float*)d_in[4]; const float* bk = (const float*)d_in[5];
  const float* wv = (const float*)d_in[6]; const float* bv = (const float*)d_in[7];
  const float* wo = (const float*)d_in[8]; const float* bo = (const float*)d_in[9];
  float* out = (float*)d_out;

  // Workspace carve-out (bf16 elements): x(4Mi) w(4Mi) q/k/vt(4Mi ea) ctx(4Mi)
  const size_t nX = (size_t)M_ * H_;       // 4 Mi
  const size_t nW = (size_t)H_ * H_;       // 1 Mi
  __bf16* x_bf  = (__bf16*)d_ws;
  __bf16* w_bf  = x_bf + nX;
  __bf16* q_ws  = w_bf + 4 * nW;
  __bf16* k_ws  = q_ws + nX;
  __bf16* vt_ws = k_ws + nX;
  __bf16* c_ws  = vt_ws + nX;

  cvt_bf16_kernel<<<dim3(nX / (8 * 256)), 256, 0, stream>>>(hs, x_bf, (int)nX);
  cvt_bf16_kernel<<<dim3(nW / (8 * 256)), 256, 0, stream>>>(wq, w_bf + 0 * nW, (int)nW);
  cvt_bf16_kernel<<<dim3(nW / (8 * 256)), 256, 0, stream>>>(wk, w_bf + 1 * nW, (int)nW);
  cvt_bf16_kernel<<<dim3(nW / (8 * 256)), 256, 0, stream>>>(wv, w_bf + 2 * nW, (int)nW);
  cvt_bf16_kernel<<<dim3(nW / (8 * 256)), 256, 0, stream>>>(wo, w_bf + 3 * nW, (int)nW);

  qkv_kernel<<<dim3(M_ / 64, H_ / 128, 3), 256, 0, stream>>>(
      x_bf, w_bf, bq, bk, bv, q_ws, k_ws, vt_ws);
  attn_kernel<<<dim3(S_ / 64, B_ * NH_), 128, 0, stream>>>(
      q_ws, k_ws, vt_ws, mask, c_ws);
  outproj_kernel<<<dim3(M_ / 64, H_ / 128), 256, 0, stream>>>(
      c_ws, w_bf + 3 * nW, bo, out);
}
